// LatentPredictor_63410897158581
// MI455X (gfx1250) — matrix operand order repaired and checked by reading, not compiled, not run on hardware
//
#include <hip/hip_runtime.h>
#include <hip/hip_bf16.h>
#include <math.h>

typedef __attribute__((ext_vector_type(16))) _Float16 v16h;
typedef __attribute__((ext_vector_type(8)))  _Float16 v8h;
typedef __attribute__((ext_vector_type(4)))  _Float16 v4h;
typedef __attribute__((ext_vector_type(8)))  float    v8f;
typedef int v4i __attribute__((vector_size(16)));

#if defined(__has_builtin)
#if __has_builtin(__builtin_amdgcn_global_load_async_to_lds_b128) && \
    __has_builtin(__builtin_amdgcn_s_wait_asynccnt)
#define USE_ASYNC 1
#endif
#endif
#ifndef USE_ASYNC
#define USE_ASYNC 0
#endif

#if USE_ASYNC
typedef __attribute__((address_space(1))) v4i* glob_v4i;
typedef __attribute__((address_space(3))) v4i* lds_v4i;
#endif

constexpr int CB  = 64;          // batch
constexpr int CT  = 302;         // seq len
constexpr int CD  = 768;         // model dim
constexpr int CH  = 8;           // heads
constexpr int CDH = 96;          // head dim
constexpr int CL  = 4;           // layers
constexpr int CW  = 10;          // window
constexpr int CM  = CB * CT;     // 19328 rows (divisible by 128)

constexpr int BM = 128, BN = 128, BK = 32;
constexpr int LDSK  = 48;        // padded LDS K-stride in halves (96B, 16B-aligned)
constexpr int NKIT  = CD / BK;   // 24 K-steps

// ---------------------------------------------------------------------------
// Weight convert + transpose: in[K=768][N=768] f32  ->  out[N=768][K=768] f16
// ---------------------------------------------------------------------------
__global__ __launch_bounds__(256) void transpose_cvt(const float* __restrict__ in,
                                                     _Float16* __restrict__ out) {
  __shared__ float tile[32][33];
  const int x  = blockIdx.x * 32 + threadIdx.x;   // input col n
  const int y0 = blockIdx.y * 32;                 // input row k base
  for (int i = threadIdx.y; i < 32; i += 8)
    tile[i][threadIdx.x] = in[(size_t)(y0 + i) * CD + x];
  __syncthreads();
  const int ox  = blockIdx.y * 32 + threadIdx.x;  // output col k
  const int oy0 = blockIdx.x * 32;                // output row n base
  for (int i = threadIdx.y; i < 32; i += 8)
    out[(size_t)(oy0 + i) * CD + ox] = (_Float16)tile[threadIdx.x][i];
}

// ---------------------------------------------------------------------------
// h(f16) = x(f32) + pos(f32)
// ---------------------------------------------------------------------------
__global__ __launch_bounds__(256) void add_pos(const float* __restrict__ x,
                                               const float* __restrict__ pos,
                                               _Float16* __restrict__ h) {
  const size_t i4   = (size_t)blockIdx.x * blockDim.x + threadIdx.x;
  const size_t base = i4 * 4;
  const int row = (int)(base / CD);
  const int t   = row % CT;
  const int dd  = (int)(base % CD);
  float4 xv = *(const float4*)(x + base);
  float4 pv = *(const float4*)(pos + (size_t)t * CD + dd);
  v4h r;
  r[0] = (_Float16)(xv.x + pv.x);
  r[1] = (_Float16)(xv.y + pv.y);
  r[2] = (_Float16)(xv.z + pv.z);
  r[3] = (_Float16)(xv.w + pv.w);
  *(v4h*)(h + base) = r;
}

// ---------------------------------------------------------------------------
// GEMM: out[M,768] = A[M,768](f16) @ W[768,768](f16,[N][K]) + bias, f32 acc.
// blockIdx.z selects one of up to three (W,bias,out) triples (fused QKV).
// Double-buffered LDS; async global->LDS staging when available.
// ---------------------------------------------------------------------------
__global__ __launch_bounds__(256) void gemm128_f16(
    const _Float16* __restrict__ A,
    const _Float16* __restrict__ W0, const _Float16* __restrict__ W1,
    const _Float16* __restrict__ W2,
    const float* __restrict__ bb0, const float* __restrict__ bb1,
    const float* __restrict__ bb2,
    _Float16* __restrict__ o0, _Float16* __restrict__ o1,
    _Float16* __restrict__ o2,
    float* __restrict__ outF, int wantF32) {
  const int z = blockIdx.z;
  const _Float16* __restrict__ Wt  = (z == 0) ? W0 : ((z == 1) ? W1 : W2);
  const float* __restrict__ bias   = (z == 0) ? bb0 : ((z == 1) ? bb1 : bb2);
  _Float16* __restrict__ outH      = (z == 0) ? o0 : ((z == 1) ? o1 : o2);

  __shared__ alignas(16) _Float16 As[2][BM * LDSK];
  __shared__ alignas(16) _Float16 Bs[2][BN * LDSK];

  const int tid   = threadIdx.x;
  const int mBase = blockIdx.y * BM;
  const int nBase = blockIdx.x * BN;

  // wave tiling: 8 waves as 2 (M) x 4 (N); each wave = 64x32 = 4x2 WMMA tiles
  const int lane = tid & 31;
  const int w    = tid >> 5;
  const int wm   = (w & 1) * 64;
  const int wn   = (w >> 1) * 32;
  const int lr   = lane & 15;
  const int hi   = lane >> 4;   // 0: lanes 0-15, 1: lanes 16-31

  const _Float16* gA = A  + (size_t)mBase * CD;   // tile row 0, k 0
  const _Float16* gB = Wt + (size_t)nBase * CD;

  // stage one 128x32 f16 tile (16B chunks; 512 chunks, 2 per thread)
  auto stage = [&](int buf, int kk) {
#pragma unroll
    for (int i = 0; i < 2; ++i) {
      const int c   = tid + i * 256;   // 0..511
      const int row = c >> 2;
      const int j   = c & 3;
      const _Float16* ga = gA + (size_t)row * CD + kk + j * 8;
      const _Float16* gb = gB + (size_t)row * CD + kk + j * 8;
      _Float16* la = &As[buf][row * LDSK + j * 8];
      _Float16* lb = &Bs[buf][row * LDSK + j * 8];
#if USE_ASYNC
      __builtin_amdgcn_global_load_async_to_lds_b128(
          (glob_v4i)ga, (lds_v4i)la, 0, 0);
      __builtin_amdgcn_global_load_async_to_lds_b128(
          (glob_v4i)gb, (lds_v4i)lb, 0, 0);
#else
      v8h ta = *(const v8h*)ga;
      v8h tb = *(const v8h*)gb;
      *(v8h*)la = ta;
      *(v8h*)lb = tb;
#endif
    }
  };

  v8f acc[4][2] = {};

  stage(0, 0);
  for (int it = 0; it < NKIT; ++it) {
    const int cur = it & 1;
    if (it + 1 < NKIT) {
      stage(cur ^ 1, (it + 1) * BK);
#if USE_ASYNC
      __builtin_amdgcn_s_wait_asynccnt(4);  // current buffer's 4 issues done
#endif
    } else {
#if USE_ASYNC
      __builtin_amdgcn_s_wait_asynccnt(0);
#endif
    }
    __syncthreads();

    // ---- fragments (ISA 16-bit WMMA layouts) ----
    v16h af[4];
#pragma unroll
    for (int mt = 0; mt < 4; mt++) {
      // A 16x32: lane row = lr; e0..7 -> K = hi*8+0..7; e8..15 -> K = hi*8+16..23
      const _Float16* ap = &As[cur][(wm + mt * 16 + lr) * LDSK + hi * 8];
      v8h lo = *(const v8h*)ap;
      v8h hh = *(const v8h*)(ap + 16);
      af[mt] = __builtin_shufflevector(lo, hh, 0, 1, 2, 3, 4, 5, 6, 7,
                                       8, 9, 10, 11, 12, 13, 14, 15);
    }
    v16h bf[2];
#pragma unroll
    for (int nt = 0; nt < 2; nt++) {
      // B 32x16: lane col = lr; e0..7 -> K = hi*8+0..7; e8..15 -> K = hi*8+16..23
      const _Float16* bp = &Bs[cur][(wn + nt * 16 + lr) * LDSK + hi * 8];
      v8h lo = *(const v8h*)bp;
      v8h hh = *(const v8h*)(bp + 16);
      bf[nt] = __builtin_shufflevector(lo, hh, 0, 1, 2, 3, 4, 5, 6, 7,
                                       8, 9, 10, 11, 12, 13, 14, 15);
    }
#pragma unroll
    for (int mt = 0; mt < 4; mt++)
#pragma unroll
      for (int nt = 0; nt < 2; nt++)
        acc[mt][nt] = __builtin_amdgcn_wmma_f32_16x16x32_f16(
            false, af[mt], false, bf[nt], (short)0, acc[mt][nt], false, false);
    __syncthreads();  // all waves done reading 'cur' before it is re-staged
  }

  // ---- epilogue: bias + store (C layout: VGPR r -> M = r + hi*8, N = lr) ----
  float bc[2];
#pragma unroll
  for (int nt = 0; nt < 2; nt++) bc[nt] = bias[nBase + wn + nt * 16 + lr];
#pragma unroll
  for (int mt = 0; mt < 4; mt++) {
    const int row = mBase + wm + mt * 16 + hi * 8;
#pragma unroll
    for (int r = 0; r < 8; r++)
#pragma unroll
      for (int nt = 0; nt < 2; nt++) {
        const size_t idx = (size_t)(row + r) * CD + nBase + wn + nt * 16 + lr;
        const float val  = acc[mt][nt][r] + bc[nt];
        if (wantF32) outF[idx] = val;
        else         outH[idx] = (_Float16)val;
      }
  }
}

// ---------------------------------------------------------------------------
// Fused causal local attention, window 10 (f16 in, f16 out, f32 math).
// One wave per (b, head, t): lanes 0..9 score, shfl softmax, 32 lanes output.
// ---------------------------------------------------------------------------
__global__ __launch_bounds__(256) void attn_local(const _Float16* __restrict__ q,
                                                  const _Float16* __restrict__ k,
                                                  const _Float16* __restrict__ v,
                                                  _Float16* __restrict__ o) {
  __shared__ float ps[8][16];
  const int lane = threadIdx.x & 31;
  const int w    = threadIdx.x >> 5;
  const int wid  = blockIdx.x * 8 + w;   // grid sized exactly: no stragglers
  const int t  = wid % CT;
  const int hh = (wid / CT) % CH;
  const int b  = wid / (CT * CH);
  const float scale = 0.10206207261596575f;  // 1/sqrt(96)

  const int  s     = t - (CW - 1) + lane;
  const bool valid = (lane < CW) && (s >= 0);
  float sc = -1e30f;
  if (valid) {
    const _Float16* qp = q + ((size_t)(b * CT + t) * CD + hh * CDH);
    const _Float16* kp = k + ((size_t)(b * CT + s) * CD + hh * CDH);
    float a = 0.f;
#pragma unroll 8
    for (int d = 0; d < CDH; d++) a += (float)qp[d] * (float)kp[d];
    sc = a * scale;
  }
  float m = sc;
  for (int off = 16; off > 0; off >>= 1) m = fmaxf(m, __shfl_xor(m, off, 32));
  float e = valid ? __expf(sc - m) : 0.f;
  float sum = e;
  for (int off = 16; off > 0; off >>= 1) sum += __shfl_xor(sum, off, 32);
  const float p = e / sum;
  if (lane < 16) ps[w][lane] = p;
  __syncthreads();

  float ob0 = 0.f, ob1 = 0.f, ob2 = 0.f;
#pragma unroll
  for (int si = 0; si < CW; si++) {
    const int ss = t - (CW - 1) + si;
    if (ss < 0) continue;
    const float pp = ps[w][si];
    const _Float16* vp = v + ((size_t)(b * CT + ss) * CD + hh * CDH);
    ob0 += pp * (float)vp[lane];
    ob1 += pp * (float)vp[lane + 32];
    ob2 += pp * (float)vp[lane + 64];
  }
  _Float16* op = o + ((size_t)(b * CT + t) * CD + hh * CDH);
  op[lane]      = (_Float16)ob0;
  op[lane + 32] = (_Float16)ob1;
  op[lane + 64] = (_Float16)ob2;
}

// ---------------------------------------------------------------------------
extern "C" void kernel_launch(void* const* d_in, const int* in_sizes, int n_in,
                              void* d_out, int out_size, void* d_ws, size_t ws_size,
                              hipStream_t stream) {
  (void)in_sizes; (void)n_in; (void)out_size; (void)ws_size;
  const float* x   = (const float*)d_in[0];
  const float* pos = (const float*)d_in[1];
  const float* Wq  = (const float*)d_in[2];
  const float* bq  = (const float*)d_in[3];
  const float* Wk  = (const float*)d_in[4];
  const float* bk  = (const float*)d_in[5];
  const float* Wv  = (const float*)d_in[6];
  const float* bv  = (const float*)d_in[7];
  const float* Wo  = (const float*)d_in[8];
  const float* bo  = (const float*)d_in[9];

  char* ws = (char*)d_ws;
  const size_t szAct = (size_t)CM * CD * sizeof(_Float16);   // 29.7 MB
  _Float16* h16 = (_Float16*)(ws);
  _Float16* q16 = (_Float16*)(ws + 1 * szAct);
  _Float16* k16 = (_Float16*)(ws + 2 * szAct);
  _Float16* v16 = (_Float16*)(ws + 3 * szAct);
  _Float16* o16 = (_Float16*)(ws + 4 * szAct);
  _Float16* Wt  = (_Float16*)(ws + 5 * szAct);               // 16 x 768x768 f16
  const size_t MS = (size_t)CD * CD;

  // One-time weight convert+transpose (16 matrices)
  {
    dim3 tb(32, 8), tg(CD / 32, CD / 32);
    const float* bases[4] = {Wq, Wk, Wv, Wo};
    for (int l = 0; l < CL; l++)
      for (int j = 0; j < 4; j++)
        transpose_cvt<<<tg, tb, 0, stream>>>(bases[j] + (size_t)l * MS,
                                             Wt + (size_t)(l * 4 + j) * MS);
  }

  add_pos<<<(CM * CD / 4) / 256, 256, 0, stream>>>(x, pos, h16);

  const dim3 gq(CD / BN, CM / BM, 3);  // (6, 151, 3)
  const dim3 go(CD / BN, CM / BM, 1);
  for (int l = 0; l < CL; l++) {
    const _Float16* WtQ = Wt + (size_t)(l * 4 + 0) * MS;
    const _Float16* WtK = Wt + (size_t)(l * 4 + 1) * MS;
    const _Float16* WtV = Wt + (size_t)(l * 4 + 2) * MS;
    const _Float16* WtO = Wt + (size_t)(l * 4 + 3) * MS;

    gemm128_f16<<<gq, 256, 0, stream>>>(h16, WtQ, WtK, WtV,
                                        bq + l * CD, bk + l * CD, bv + l * CD,
                                        q16, k16, v16, (float*)nullptr, 0);

    attn_local<<<(CB * CH * CT) / 8, 256, 0, stream>>>(q16, k16, v16, o16);

    const int last = (l == CL - 1);
    gemm128_f16<<<go, 256, 0, stream>>>(o16, WtO, WtO, WtO,
                                        bo + l * CD, bo + l * CD, bo + l * CD,
                                        h16, h16, h16, (float*)d_out, last);
  }
}
